// RAFTPose_11871289606480
// MI455X (gfx1250) — matrix operand order repaired and checked by reading, not compiled, not run on hardware
//
#include <hip/hip_runtime.h>

// ---------------- types ----------------
typedef __bf16 bf16_t;
typedef __attribute__((ext_vector_type(8)))  bf16_t v8bf;
typedef __attribute__((ext_vector_type(16))) bf16_t v16bf;
typedef __attribute__((ext_vector_type(8)))  float  v8f;

// ---------------- problem constants ----------------
constexpr int kC = 256, kH = 48, kW = 64, kHW = kH * kW;          // 3072
constexpr int kHID = 128, kCTX = 64;
constexpr int kLEVELS = 4, kRADIUS = 4, kP = 81;
constexpr int kCORRD = kP * kLEVELS;                               // 324
constexpr int kNS = 37, kTOPK = 3, kITERS = 4;
constexpr int kGIN = kHID + kCORRD + kCTX;                         // 516
constexpr int kK9 = kGIN * 9;                                      // 4644
constexpr int kK9P = 4672;                                         // 146*32
constexpr int kKH = kHID * 9;                                      // 1152 = 36*32

// ---------------- small math helpers ----------------
__device__ __forceinline__ void qmul4(const float* a, const float* b, float* o) {
  o[0] = a[0]*b[0] - a[1]*b[1] - a[2]*b[2] - a[3]*b[3];
  o[1] = a[0]*b[1] + a[1]*b[0] + a[2]*b[3] - a[3]*b[2];
  o[2] = a[0]*b[2] - a[1]*b[3] + a[2]*b[0] + a[3]*b[1];
  o[3] = a[0]*b[3] + a[1]*b[2] - a[2]*b[1] + a[3]*b[0];
}
__device__ __forceinline__ void qapply3(const float* q, const float* v, float* o) {
  float qv[4] = {0.f, v[0], v[1], v[2]};
  float qc[4] = {q[0], -q[1], -q[2], -q[3]};
  float t1[4], t2[4];
  qmul4(q, qv, t1);
  qmul4(t1, qc, t2);
  o[0] = t2[1]; o[1] = t2[2]; o[2] = t2[3];
}
__device__ __forceinline__ float bilin(const float* __restrict__ row, int Hl, int Wl,
                                       float x, float y) {
  float x0f = floorf(x), y0f = floorf(y);
  int x0 = (int)x0f, y0 = (int)y0f;
  float ax = x - x0f, ay = y - y0f;
  float r = 0.f;
  if (y0 >= 0 && y0 < Hl) {
    const float* r0 = row + (size_t)y0 * Wl;
    if (x0 >= 0 && x0 < Wl)         r += (1.f - ax) * (1.f - ay) * r0[x0];
    if (x0 + 1 >= 0 && x0 + 1 < Wl) r += ax * (1.f - ay) * r0[x0 + 1];
  }
  if (y0 + 1 >= 0 && y0 + 1 < Hl) {
    const float* r1 = row + (size_t)(y0 + 1) * Wl;
    if (x0 >= 0 && x0 < Wl)         r += (1.f - ax) * ay * r1[x0];
    if (x0 + 1 >= 0 && x0 + 1 < Wl) r += ax * ay * r1[x0 + 1];
  }
  return r;
}
__device__ __forceinline__ void projectPix(int p, float d, const float* __restrict__ R,
                                           const float* __restrict__ t, float fx, float fy,
                                           float cx, float cy, float& up, float& vp) {
  float u = (float)(p & 63);   // W = 64
  float v = (float)(p >> 6);
  float X = (u - cx) / fx * d;
  float Y = (v - cy) / fy * d;
  float Z = d;
  float xc = R[0]*X + R[1]*Y + R[2]*Z + t[0];
  float yc = R[3]*X + R[4]*Y + R[5]*Z + t[1];
  float zc = R[6]*X + R[7]*Y + R[8]*Z + t[2];
  zc = fmaxf(zc, 0.1f);
  up = fx * xc / zc + cx;
  vp = fy * yc / zc + cy;
}

__device__ const float kBaseR[12][3] = {
  {0,0,0},{0,0,0},{0,0,0},{0,0,0},{0,0,0},{0,0,0},
  {1,0,0},{-1,0,0},{0,1,0},{0,-1,0},{0,0,1},{0,0,-1}};
__device__ const float kBaseT[12][3] = {
  {1,0,0},{-1,0,0},{0,1,0},{0,-1,0},{0,0,1},{0,0,-1},
  {0,0,0},{0,0,0},{0,0,0},{0,0,0},{0,0,0},{0,0,0}};

// ---------------- conversion / init kernels ----------------
// (C,HW) f32 -> (HW,C) bf16 transpose; used for both feature maps
__global__ void cvt_T_kernel(const float* __restrict__ src, bf16_t* __restrict__ dst) {
  size_t idx = (size_t)blockIdx.x * blockDim.x + threadIdx.x;   // < HW*C
  int i = (int)(idx / kC);
  int c = (int)(idx - (size_t)i * kC);
  dst[idx] = (bf16_t)src[(size_t)c * kHW + i];
}
// pad conv weights (M x Kin) -> (M x Kpad) bf16, zeros in pad
__global__ void cvt_w_kernel(const float* __restrict__ w, bf16_t* __restrict__ dst,
                             int Kin, int Kpad) {
  size_t idx = (size_t)blockIdx.x * blockDim.x + threadIdx.x;   // < M*Kpad
  int m = (int)(idx / Kpad);
  int k = (int)(idx - (size_t)m * Kpad);
  dst[idx] = (k < Kin) ? (bf16_t)w[(size_t)m * Kin + k] : (bf16_t)0.f;
}
// zero h / h-slot of hx, copy context into ctx-slot of hx (bf16)
__global__ void init_state_kernel(const float* __restrict__ ctx, float* __restrict__ h,
                                  bf16_t* __restrict__ hx) {
  size_t idx = (size_t)blockIdx.x * blockDim.x + threadIdx.x;   // < (HID+CTX)*HW
  if (idx < (size_t)kHID * kHW) {
    h[idx] = 0.f;
    hx[idx] = (bf16_t)0.f;
  } else {
    size_t r = idx - (size_t)kHID * kHW;
    int c = (int)(r / kHW), p = (int)(r - (size_t)c * kHW);
    hx[(size_t)(kHID + kCORRD + c) * kHW + p] = (bf16_t)ctx[r];
  }
}
__global__ void init_misc_kernel(const float* __restrict__ intr, float* __restrict__ pose,
                                 float* __restrict__ cam) {
  if (threadIdx.x == 0) {
    pose[0] = 1.f; pose[1] = 0.f; pose[2] = 0.f; pose[3] = 0.f;
    pose[4] = 0.f; pose[5] = 0.f; pose[6] = 0.f;
    cam[0] = 60.f + 40.f * intr[0];
    cam[1] = 60.f + 40.f * intr[1];
    cam[2] = 32.f + 4.f * (intr[2] - 0.5f);
    cam[3] = 24.f + 4.f * (intr[3] - 0.5f);
  }
}

// ---------------- WMMA bf16 GEMM ----------------
// D(MxN) = A(MxK) @ BT(NxK)^T * scale.  Both operands K-contiguous.
// N fixed at 3072, K compile-time -> all strides become instruction immediates.
// One wave per 16x16 tile; 8 waves/block cover 128 rows of M; all waves of a
// block share the same 16-col B panel (L2 reuse).
template <int K>
__global__ __launch_bounds__(256) void wmma_gemm_kernel(
    const bf16_t* __restrict__ A, const bf16_t* __restrict__ BT,
    float* __restrict__ D, float scale) {
  constexpr int N = kHW;  // 3072
  int wave = threadIdx.x >> 5;
  int lane = threadIdx.x & 31;
  int tileM = blockIdx.y * 128 + wave * 16;
  int tileN = blockIdx.x * 16;
  int half = lane >> 4;      // 0: lanes 0-15, 1: lanes 16-31
  int lrow = lane & 15;

  // A layout (ISA 7.12.2, 16-bit A 16x32): lanes0-15 K{0-7,16-23}, lanes16-31 K{8-15,24-31}
  const bf16_t* arow = A + (size_t)(tileM + lrow) * K + 8 * half;
  // B layout: lanes0-15 col N=lrow holds K{0-7,16-23}; lanes16-31 same cols K{8-15,24-31}
  const bf16_t* brow = BT + (size_t)(tileN + lrow) * K + 8 * half;

  v8f acc;
#pragma unroll
  for (int i = 0; i < 8; ++i) acc[i] = 0.f;

#pragma unroll 2
  for (int k0 = 0; k0 < K; k0 += 32) {
    v8bf a0 = *reinterpret_cast<const v8bf*>(arow + k0);
    v8bf a1 = *reinterpret_cast<const v8bf*>(arow + k0 + 16);
    v16bf av;
#pragma unroll
    for (int j = 0; j < 8; ++j) { av[j] = a0[j]; av[8 + j] = a1[j]; }
    v8bf b0 = *reinterpret_cast<const v8bf*>(brow + k0);
    v8bf b1 = *reinterpret_cast<const v8bf*>(brow + k0 + 16);
    v16bf bv;
#pragma unroll
    for (int j = 0; j < 8; ++j) { bv[j] = b0[j]; bv[8 + j] = b1[j]; }
    __builtin_prefetch(arow + k0 + 32, 0, 1);                // global_prefetch_b8
    __builtin_prefetch(brow + k0 + 32, 0, 1);
    acc = __builtin_amdgcn_wmma_f32_16x16x32_bf16(
        false, av, false, bv, (short)0, acc, false, false);
  }
  // C/D layout: lanes0-15: N=lane, M=vgpr; lanes16-31: N=lane-16, M=8+vgpr
  float* drow = D + (size_t)(tileM + 8 * half) * N + tileN + lrow;
#pragma unroll
  for (int r = 0; r < 8; ++r) drow[r * N] = acc[r] * scale;
}

// ---------------- pyramid avg-pool ----------------
__global__ void avgpool_kernel(const float* __restrict__ src, float* __restrict__ dst,
                               int Hl, int Wl) {
  int H2 = Hl >> 1, W2 = Wl >> 1, per = H2 * W2;
  size_t idx = (size_t)blockIdx.x * blockDim.x + threadIdx.x;   // < HW*per
  int j = (int)(idx % per);
  size_t p = idx / per;
  int y = j / W2, x = j - y * W2;
  const float* ra = src + p * (size_t)(Hl * Wl) + (size_t)(2 * y) * Wl + 2 * x;
  const float* rb = ra + Wl;
  dst[p * (size_t)per + j] = 0.25f * (ra[0] + ra[1] + rb[0] + rb[1]);
}

// ---------------- per-iteration kernels ----------------
__global__ void pose_expand_kernel(const float* __restrict__ pose, float* __restrict__ Rm,
                                   float* __restrict__ Tt, float* __restrict__ conf) {
  int n = threadIdx.x;
  if (n >= kNS) return;
  conf[n] = 0.f;
  float dq[4], dt[3];
  if (n < 36) {
    int b = n / 3, si = n - b * 3;
    float sc = (si == 0) ? 0.25f : ((si == 1) ? 1.f : 4.f);
    float rx = kBaseR[b][0], ry = kBaseR[b][1], rz = kBaseR[b][2];
    float nr = sqrtf(rx * rx + ry * ry + rz * rz);
    float inv = 1.f / fmaxf(nr, 1e-8f);
    float ha = 0.02f * sc * 0.5f;
    float sh = sinf(ha);
    dq[0] = cosf(ha); dq[1] = rx * inv * sh; dq[2] = ry * inv * sh; dq[3] = rz * inv * sh;
    float ts = 0.02f * sc;
    dt[0] = kBaseT[b][0] * ts; dt[1] = kBaseT[b][1] * ts; dt[2] = kBaseT[b][2] * ts;
  } else {
    dq[0] = 1.f; dq[1] = dq[2] = dq[3] = 0.f;
    dt[0] = dt[1] = dt[2] = 0.f;
  }
  float qc[4] = {pose[0], pose[1], pose[2], pose[3]};
  float q[4];
  qmul4(qc, dq, q);
  float inq = rsqrtf(q[0]*q[0] + q[1]*q[1] + q[2]*q[2] + q[3]*q[3]);
  q[0] *= inq; q[1] *= inq; q[2] *= inq; q[3] *= inq;
  float ta[3];
  qapply3(qc, dt, ta);
  Tt[n * 3 + 0] = pose[4] + ta[0];
  Tt[n * 3 + 1] = pose[5] + ta[1];
  Tt[n * 3 + 2] = pose[6] + ta[2];
  float w = q[0], x = q[1], y = q[2], z = q[3];
  float* R = Rm + n * 9;
  R[0] = 1.f - 2.f * (y * y + z * z); R[1] = 2.f * (x * y - w * z); R[2] = 2.f * (x * z + w * y);
  R[3] = 2.f * (x * y + w * z); R[4] = 1.f - 2.f * (x * x + z * z); R[5] = 2.f * (y * z - w * x);
  R[6] = 2.f * (x * z - w * y); R[7] = 2.f * (y * z + w * x); R[8] = 1.f - 2.f * (x * x + y * y);
}

__global__ __launch_bounds__(128) void conf_kernel(
    const float* __restrict__ depth, const float* __restrict__ cam,
    const float* __restrict__ Rm, const float* __restrict__ Tt,
    const float* __restrict__ c0, const float* __restrict__ c1,
    const float* __restrict__ c2, const float* __restrict__ c3,
    float* __restrict__ conf) {
  int n = blockIdx.x;
  int p = blockIdx.y * 128 + threadIdx.x;
  float fx = cam[0], fy = cam[1], cx = cam[2], cy = cam[3];
  float d = 4.f + 20.f * depth[p];
  float up, vp;
  projectPix(p, d, Rm + n * 9, Tt + n * 3, fx, fy, cx, cy, up, vp);

  const float* rows[4] = {c0 + (size_t)p * 3072, c1 + (size_t)p * 768,
                          c2 + (size_t)p * 192,  c3 + (size_t)p * 48};
  const int lh[4] = {48, 24, 12, 6};
  const int lw[4] = {64, 32, 16, 8};
  float sum = 0.f, sc = 1.f;
  for (int l = 0; l < kLEVELS; ++l) {
    float ul = up / sc, vl = vp / sc;
    for (int oy = -kRADIUS; oy <= kRADIUS; ++oy)
      for (int ox = -kRADIUS; ox <= kRADIUS; ++ox)
        sum += bilin(rows[l], lh[l], lw[l], ul + (float)ox, vl + (float)oy);
    sc *= 2.f;
  }
  __shared__ float red[128];
  red[threadIdx.x] = sum;
  __syncthreads();
  for (int s = 64; s > 0; s >>= 1) {
    if (threadIdx.x < s) red[threadIdx.x] += red[threadIdx.x + s];
    __syncthreads();
  }
  if (threadIdx.x == 0) atomicAdd(&conf[n], red[0]);
}

__global__ void topk_kernel(const float* __restrict__ conf, float* __restrict__ wk,
                            int* __restrict__ topi) {
  if (threadIdx.x != 0) return;
  const float inv = 1.f / ((float)kHW * (float)kCORRD);
  float v[kNS];
  for (int n = 0; n < kNS; ++n) v[n] = conf[n] * inv;
  int pick[kTOPK]; float pv[kTOPK];
  for (int k = 0; k < kTOPK; ++k) {
    int bi = -1; float bv = -3.4e38f;
    for (int n = 0; n < kNS; ++n) {
      bool used = false;
      for (int j = 0; j < k; ++j) if (pick[j] == n) used = true;
      if (!used && v[n] > bv) { bv = v[n]; bi = n; }
    }
    pick[k] = bi; pv[k] = bv;
  }
  float m = pv[0];
  float e0 = expf(pv[0] - m), e1 = expf(pv[1] - m), e2 = expf(pv[2] - m);
  float s = e0 + e1 + e2;
  wk[0] = e0 / s; wk[1] = e1 / s; wk[2] = e2 / s;
  topi[0] = pick[0]; topi[1] = pick[1]; topi[2] = pick[2];
}

__global__ __launch_bounds__(128) void fused_kernel(
    const float* __restrict__ depth, const float* __restrict__ cam,
    const float* __restrict__ Rm, const float* __restrict__ Tt,
    const float* __restrict__ c0, const float* __restrict__ c1,
    const float* __restrict__ c2, const float* __restrict__ c3,
    const float* __restrict__ wk, const int* __restrict__ topi,
    bf16_t* __restrict__ hx) {
  int p = blockIdx.x * 128 + threadIdx.x;
  float fx = cam[0], fy = cam[1], cx = cam[2], cy = cam[3];
  float d = 4.f + 20.f * depth[p];
  float uk[kTOPK], vk[kTOPK], wkv[kTOPK];
  for (int k = 0; k < kTOPK; ++k) {
    int n = topi[k];
    wkv[k] = wk[k];
    projectPix(p, d, Rm + n * 9, Tt + n * 3, fx, fy, cx, cy, uk[k], vk[k]);
  }
  const float* rows[4] = {c0 + (size_t)p * 3072, c1 + (size_t)p * 768,
                          c2 + (size_t)p * 192,  c3 + (size_t)p * 48};
  const int lh[4] = {48, 24, 12, 6};
  const int lw[4] = {64, 32, 16, 8};
  float sc = 1.f;
  for (int l = 0; l < kLEVELS; ++l) {
    float ul[kTOPK], vl[kTOPK];
    for (int k = 0; k < kTOPK; ++k) { ul[k] = uk[k] / sc; vl[k] = vk[k] / sc; }
    for (int oy = -kRADIUS; oy <= kRADIUS; ++oy) {
      for (int ox = -kRADIUS; ox <= kRADIUS; ++ox) {
        float val = 0.f;
        for (int k = 0; k < kTOPK; ++k)
          val += wkv[k] * bilin(rows[l], lh[l], lw[l], ul[k] + (float)ox, vl[k] + (float)oy);
        int ch = kHID + l * kP + (oy + kRADIUS) * 9 + (ox + kRADIUS);
        hx[(size_t)ch * kHW + p] = (bf16_t)val;
      }
    }
    sc *= 2.f;
  }
}

// ---------------- im2col (3x3, SAME) into (HW x Kpad) K-contiguous patch ----------------
__global__ void im2col_kernel(const bf16_t* __restrict__ src, bf16_t* __restrict__ dst,
                              int Cin, int Kpad) {
  size_t idx = (size_t)blockIdx.x * blockDim.x + threadIdx.x;   // < HW*Kpad
  int r = (int)(idx % Kpad);
  int p = (int)(idx / Kpad);
  bf16_t val = (bf16_t)0.f;
  if (r < Cin * 9) {
    int c = r / 9, o = r - c * 9;
    int ky = o / 3, kx = o - ky * 3;
    int py = (p >> 6) + ky - 1;
    int px = (p & 63) + kx - 1;
    if ((unsigned)px < 64u && (unsigned)py < 48u)
      val = src[(size_t)c * kHW + py * 64 + px];
  }
  dst[idx] = val;
}

// ---------------- pointwise GRU ops ----------------
__global__ void pw_zr_kernel(const float* __restrict__ zpre, const float* __restrict__ rpre,
                             const float* __restrict__ bz, const float* __restrict__ br,
                             const float* __restrict__ h, float* __restrict__ zg,
                             bf16_t* __restrict__ hx) {
  size_t idx = (size_t)blockIdx.x * blockDim.x + threadIdx.x;   // < HID*HW
  int m = (int)(idx / kHW);
  float z = 1.f / (1.f + expf(-(zpre[idx] + bz[m])));
  float r = 1.f / (1.f + expf(-(rpre[idx] + br[m])));
  zg[idx] = z;
  hx[idx] = (bf16_t)(r * h[idx]);   // q-conv input slot
}
__global__ void pw_q_kernel(const float* __restrict__ qpre, const float* __restrict__ bq,
                            const float* __restrict__ zg, float* __restrict__ h,
                            bf16_t* __restrict__ hx) {
  size_t idx = (size_t)blockIdx.x * blockDim.x + threadIdx.x;
  int m = (int)(idx / kHW);
  float q = tanhf(qpre[idx] + bq[m]);
  float z = zg[idx];
  float hn = (1.f - z) * h[idx] + z * q;
  h[idx] = hn;
  hx[idx] = (bf16_t)hn;             // new h back into hx slot
}

__global__ __launch_bounds__(256) void hfeat_kernel(const float* __restrict__ hout,
                                                    const float* __restrict__ bh,
                                                    float* __restrict__ feat) {
  int m = blockIdx.x;
  float s = 0.f;
  for (int p = threadIdx.x; p < kHW; p += 256)
    s += fmaxf(hout[(size_t)m * kHW + p] + bh[m], 0.f);
  __shared__ float red[256];
  red[threadIdx.x] = s;
  __syncthreads();
  for (int st = 128; st > 0; st >>= 1) {
    if (threadIdx.x < st) red[threadIdx.x] += red[threadIdx.x + st];
    __syncthreads();
  }
  if (threadIdx.x == 0) feat[m] = red[0] * (1.f / (float)kHW);
}

__global__ void pose_update_kernel(const float* __restrict__ feat,
                                   const float* __restrict__ Wfc,
                                   const float* __restrict__ bfc,
                                   float* __restrict__ pose) {
  if (threadIdx.x != 0) return;
  float delta[7];
  for (int j = 0; j < 7; ++j) {
    float s = bfc[j];
    for (int m = 0; m < kHID; ++m) s += feat[m] * Wfc[m * 7 + j];
    delta[j] = 0.01f * s;
  }
  float qc[4] = {pose[0], pose[1], pose[2], pose[3]};
  float tc[3] = {pose[4], pose[5], pose[6]};
  float dq[4] = {1.f + delta[0], delta[1], delta[2], delta[3]};
  float inv = rsqrtf(dq[0]*dq[0] + dq[1]*dq[1] + dq[2]*dq[2] + dq[3]*dq[3]);
  dq[0] *= inv; dq[1] *= inv; dq[2] *= inv; dq[3] *= inv;
  float ta[3];
  qapply3(qc, &delta[4], ta);
  float qn[4];
  qmul4(qc, dq, qn);
  float in2 = rsqrtf(qn[0]*qn[0] + qn[1]*qn[1] + qn[2]*qn[2] + qn[3]*qn[3]);
  pose[0] = qn[0] * in2; pose[1] = qn[1] * in2; pose[2] = qn[2] * in2; pose[3] = qn[3] * in2;
  pose[4] = tc[0] + ta[0]; pose[5] = tc[1] + ta[1]; pose[6] = tc[2] + ta[2];
}

__global__ void copy_out_kernel(const float* __restrict__ pose, float* __restrict__ out) {
  if (threadIdx.x < 7) out[threadIdx.x] = pose[threadIdx.x];
}

// ---------------- launch ----------------
extern "C" void kernel_launch(void* const* d_in, const int* in_sizes, int n_in,
                              void* d_out, int out_size, void* d_ws, size_t ws_size,
                              hipStream_t stream) {
  (void)in_sizes; (void)n_in; (void)out_size; (void)ws_size;
  const float* f_rgb = (const float*)d_in[0];
  const float* f_dep = (const float*)d_in[1];
  const float* depth = (const float*)d_in[2];
  const float* ctx   = (const float*)d_in[3];
  const float* intr  = (const float*)d_in[4];
  const float* Wz = (const float*)d_in[5];  const float* bz = (const float*)d_in[6];
  const float* Wr = (const float*)d_in[7];  const float* br = (const float*)d_in[8];
  const float* Wq = (const float*)d_in[9];  const float* bq = (const float*)d_in[10];
  const float* Wh = (const float*)d_in[11]; const float* bh = (const float*)d_in[12];
  const float* Wfc = (const float*)d_in[13]; const float* bfc = (const float*)d_in[14];

  char* base = (char*)d_ws;
  size_t off = 0;
  auto alloc = [&](size_t bytes) -> char* {
    char* p = base + off;
    off += (bytes + 255) & ~(size_t)255;
    return p;
  };
  float*  corr0 = (float*) alloc((size_t)kHW * 3072 * 4);
  float*  corr1 = (float*) alloc((size_t)kHW * 768 * 4);
  float*  corr2 = (float*) alloc((size_t)kHW * 192 * 4);
  float*  corr3 = (float*) alloc((size_t)kHW * 48 * 4);
  bf16_t* fdT   = (bf16_t*)alloc((size_t)kHW * kC * 2);     // A: (HW x C)
  bf16_t* rgbT  = (bf16_t*)alloc((size_t)kHW * kC * 2);     // BT: (HW x C)
  bf16_t* hx    = (bf16_t*)alloc((size_t)kGIN * kHW * 2);
  bf16_t* patch = (bf16_t*)alloc((size_t)kHW * kK9P * 2);   // BT: (HW x Kpad)
  bf16_t* wzb   = (bf16_t*)alloc((size_t)kHID * kK9P * 2);
  bf16_t* wrb   = (bf16_t*)alloc((size_t)kHID * kK9P * 2);
  bf16_t* wqb   = (bf16_t*)alloc((size_t)kHID * kK9P * 2);
  bf16_t* whb   = (bf16_t*)alloc((size_t)kHID * kKH * 2);
  float*  hbuf  = (float*) alloc((size_t)kHID * kHW * 4);
  float*  zg    = (float*) alloc((size_t)kHID * kHW * 4);
  float*  gA    = (float*) alloc((size_t)kHID * kHW * 4);
  float*  gB    = (float*) alloc((size_t)kHID * kHW * 4);
  float*  pose  = (float*) alloc(64);
  float*  cam   = (float*) alloc(64);
  float*  Rm    = (float*) alloc(kNS * 9 * 4);
  float*  Tt    = (float*) alloc(kNS * 3 * 4);
  float*  confb = (float*) alloc(kNS * 4);
  float*  wkb   = (float*) alloc(64);
  int*    topib = (int*)   alloc(64);
  float*  featb = (float*) alloc(kHID * 4);

  // ---- precompute: bf16 transposes, weights, init state, correlation GEMM, pyramid ----
  cvt_T_kernel<<<(kHW * kC) / 256, 256, 0, stream>>>(f_rgb, rgbT);
  cvt_T_kernel<<<(kHW * kC) / 256, 256, 0, stream>>>(f_dep, fdT);
  cvt_w_kernel<<<(kHID * kK9P) / 256, 256, 0, stream>>>(Wz, wzb, kK9, kK9P);
  cvt_w_kernel<<<(kHID * kK9P) / 256, 256, 0, stream>>>(Wr, wrb, kK9, kK9P);
  cvt_w_kernel<<<(kHID * kK9P) / 256, 256, 0, stream>>>(Wq, wqb, kK9, kK9P);
  cvt_w_kernel<<<(kHID * kKH) / 256, 256, 0, stream>>>(Wh, whb, kKH, kKH);
  init_state_kernel<<<((kHID + kCTX) * kHW) / 256, 256, 0, stream>>>(ctx, hbuf, hx);
  init_misc_kernel<<<1, 32, 0, stream>>>(intr, pose, cam);

  // corr = fdT(3072x256) @ rgbT(3072x256)^T / 16
  wmma_gemm_kernel<kC><<<dim3(kHW / 16, kHW / 128), 256, 0, stream>>>(
      fdT, rgbT, corr0, 0.0625f);
  avgpool_kernel<<<(kHW * 768) / 256, 256, 0, stream>>>(corr0, corr1, 48, 64);
  avgpool_kernel<<<(kHW * 192) / 256, 256, 0, stream>>>(corr1, corr2, 24, 32);
  avgpool_kernel<<<(kHW * 48) / 256, 256, 0, stream>>>(corr2, corr3, 12, 16);

  // ---- iterations ----
  for (int it = 0; it < kITERS; ++it) {
    pose_expand_kernel<<<1, 64, 0, stream>>>(pose, Rm, Tt, confb);
    conf_kernel<<<dim3(kNS, kHW / 128), 128, 0, stream>>>(
        depth, cam, Rm, Tt, corr0, corr1, corr2, corr3, confb);
    topk_kernel<<<1, 32, 0, stream>>>(confb, wkb, topib);
    fused_kernel<<<kHW / 128, 128, 0, stream>>>(
        depth, cam, Rm, Tt, corr0, corr1, corr2, corr3, wkb, topib, hx);

    // z and r gates
    im2col_kernel<<<(kHW * kK9P) / 256, 256, 0, stream>>>(hx, patch, kGIN, kK9P);
    wmma_gemm_kernel<kK9P><<<dim3(kHW / 16, 1), 256, 0, stream>>>(wzb, patch, gA, 1.f);
    wmma_gemm_kernel<kK9P><<<dim3(kHW / 16, 1), 256, 0, stream>>>(wrb, patch, gB, 1.f);
    pw_zr_kernel<<<(kHID * kHW) / 256, 256, 0, stream>>>(gA, gB, bz, br, hbuf, zg, hx);

    // q gate (hx[0:128] now holds r*h)
    im2col_kernel<<<(kHW * kK9P) / 256, 256, 0, stream>>>(hx, patch, kGIN, kK9P);
    wmma_gemm_kernel<kK9P><<<dim3(kHW / 16, 1), 256, 0, stream>>>(wqb, patch, gA, 1.f);
    pw_q_kernel<<<(kHID * kHW) / 256, 256, 0, stream>>>(gA, bq, zg, hbuf, hx);

    // Wh conv on new h, relu, spatial mean, pose update
    im2col_kernel<<<(kHW * kKH) / 256, 256, 0, stream>>>(hx, patch, kHID, kKH);
    wmma_gemm_kernel<kKH><<<dim3(kHW / 16, 1), 256, 0, stream>>>(whb, patch, gA, 1.f);
    hfeat_kernel<<<kHID, 256, 0, stream>>>(gA, bh, featb);
    pose_update_kernel<<<1, 32, 0, stream>>>(featb, Wfc, bfc, pose);
  }

  copy_out_kernel<<<1, 32, 0, stream>>>(pose, (float*)d_out);
}
